// DeformResBlock_78151224918568
// MI455X (gfx1250) — compile-verified, matrix operand order checked by reading, not run on hardware
//
#include <hip/hip_runtime.h>
#include <hip/hip_bf16.h>
#include <stddef.h>

// ---------------------------------------------------------------------------
// DeformResBlock for MI455X (gfx1250, wave32, WMMA).
//
// The three big contractions (DCN1 einsum, DCN2 einsum, final conv) are each
// M=65536, N=64, K=576 GEMMs -> run on v_wmma_f32_16x16x32_f16 with f32
// accumulation. Everything else (offset conv, bilinear gather) is bandwidth-
// trivial (all tensors fit in the 192MB L2) and runs as VALU prep stages.
// ---------------------------------------------------------------------------

#define Bn   4
#define Cn   64
#define OCn  32
#define Hn   128
#define Wn   128
#define HW   (Hn * Wn)
#define K9   9
#define KDIM (K9 * Cn)        // 576
#define KTILES (KDIM / 32)    // 18 wmma K-steps
#define MT   16               // pixel tile (M)
#define ACT_STRIDE 584        // 576 + 8 halfs pad (keeps 16B alignment, breaks bank pattern)

typedef _Float16 v16h __attribute__((ext_vector_type(16)));
typedef _Float16 v8h  __attribute__((ext_vector_type(8)));
typedef float    v8f  __attribute__((ext_vector_type(8)));

// ---------------------------------------------------------------------------
// Kernel 1: 3x3 offset conv (32ch -> 27ch) + split into y-offset / x-offset /
// sigmoid(mask) planes. One thread per pixel; weights staged in LDS (31KB).
// ---------------------------------------------------------------------------
__global__ __launch_bounds__(256) void offset_conv_kernel(
    const float* __restrict__ off_feat, const float* __restrict__ w_off,
    const float* __restrict__ b_off,
    float* __restrict__ ybuf, float* __restrict__ xbuf, float* __restrict__ mbuf)
{
    __shared__ float wl[27 * OCn * 9];     // 7776 floats = 31104 B
    const int tid = threadIdx.x;
    for (int i = tid; i < 27 * OCn * 9; i += 256) wl[i] = w_off[i];
    __syncthreads();

    const int gid = blockIdx.x * 256 + tid;        // 0 .. B*H*W-1
    const int w = gid % Wn;
    const int h = (gid / Wn) % Hn;
    const int b = gid / HW;

    float acc[27];
#pragma unroll
    for (int t = 0; t < 27; ++t) acc[t] = b_off[t];

    for (int ci = 0; ci < OCn; ++ci) {
        const float* fp = off_feat + (size_t)(b * OCn + ci) * HW;
        float vals[9];
#pragma unroll
        for (int k = 0; k < 9; ++k) {
            const int y = h + k / 3 - 1;
            const int x = w + k % 3 - 1;
            const bool ok = (y >= 0) && (y < Hn) && (x >= 0) && (x < Wn);
            vals[k] = ok ? fp[y * Wn + x] : 0.0f;
        }
#pragma unroll
        for (int t = 0; t < 27; ++t) {
            const float* wp = &wl[(t * OCn + ci) * 9];
#pragma unroll
            for (int k = 0; k < 9; ++k) acc[t] += wp[k] * vals[k];
        }
    }

    const int pix = h * Wn + w;
#pragma unroll
    for (int kk = 0; kk < 9; ++kk) {
        const int o = (b * 9 + kk) * HW + pix;
        ybuf[o] = acc[2 * kk];                       // om channel 2k  -> off_y
        xbuf[o] = acc[2 * kk + 1];                   // om channel 2k+1-> off_x
        mbuf[o] = 1.0f / (1.0f + __expf(-acc[18 + kk]));  // sigmoid mask
    }
}

// ---------------------------------------------------------------------------
// Kernel 2: pack (O=64, I=64, 3, 3) fp32 weights into f16 fragment-major
// B-layout: Bp[nchunk][ktile][lane][16 halfs]; each lane's 16 halfs are a
// contiguous 32B chunk -> global_load_b128 x2 per fragment in the GEMM.
// K index convention: K = k9*64 + i  (kernel tap major, channel minor).
// ---------------------------------------------------------------------------
__global__ __launch_bounds__(256) void pack_w_kernel(
    const float* __restrict__ w, _Float16* __restrict__ Bp)
{
    const int t = blockIdx.x * 256 + threadIdx.x;    // 4*18*32*16 = 36864
    if (t >= (Cn / 16) * KTILES * 32 * 16) return;
    const int elem = t & 15;
    const int lane = (t >> 4) & 31;
    const int kt   = (t >> 9) % KTILES;
    const int nc   = t / (512 * KTILES);
    const int n    = nc * 16 + (lane & 15);
    const int hi   = (lane >> 4) & 1;
    // B fragment: elems 0..7 -> K = kt*32 + (hi?8:0)+e ; elems 8..15 -> +16
    const int K = kt * 32 + ((elem < 8) ? ((hi ? 8 : 0) + elem)
                                        : ((hi ? 24 : 16) + (elem - 8)));
    const int k9 = K >> 6;          // tap 0..8
    const int i  = K & 63;          // input channel
    Bp[t] = (_Float16)w[(size_t)(n * Cn + i) * 9 + k9];
}

// ---------------------------------------------------------------------------
// Shared WMMA GEMM phase: act (16 x 576 f16 tile in LDS) x packed weights.
// A fragment layout (16x16x32 f16): lane L holds row M=L%16; lanes<16 hold
// K = {0..7,16..23}, lanes>=16 hold K = {8..15,24..31} of each 32-K tile.
// ---------------------------------------------------------------------------
__device__ inline v8f wmma_gemm_576(const _Float16* act, const _Float16* bp,
                                    int lane)
{
    const int row = lane & 15;
    const int hi  = (lane >> 4) & 1;
    __builtin_prefetch(bp, 0, 1);   // global_prefetch for the weight stream
    v8f acc{};
#pragma unroll
    for (int kt = 0; kt < KTILES; ++kt) {
        const _Float16* ap = &act[row * ACT_STRIDE + kt * 32 + hi * 8];
        v8h alo = *(const v8h*)(ap);          // K base + 0..7
        v8h ahi = *(const v8h*)(ap + 16);     // K base + 16..23
        v16h afrag = __builtin_shufflevector(alo, ahi,
                         0, 1, 2, 3, 4, 5, 6, 7, 8, 9, 10, 11, 12, 13, 14, 15);
        v16h bfrag = *(const v16h*)(bp + ((size_t)kt * 32 + lane) * 16);
        acc = __builtin_amdgcn_wmma_f32_16x16x32_f16(
                  false, afrag, false, bfrag, (short)0, acc, false, false);
    }
    return acc;
}

// ---------------------------------------------------------------------------
// Kernel 3: DCN main kernel. One WG = 16 consecutive pixels of one row,
// 128 threads (4 waves; wave w computes output channels [16w,16w+16)).
// MODE 0: leaky-relu epilogue (first DCN). MODE 1: +residual (second DCN).
// ---------------------------------------------------------------------------
template <int MODE>
__global__ __launch_bounds__(128) void dcn_gemm_kernel(
    const float* __restrict__ x,
    const float* __restrict__ ybuf, const float* __restrict__ xbuf,
    const float* __restrict__ mbuf,
    const _Float16* __restrict__ Bp, const float* __restrict__ bias,
    const float* __restrict__ resid, float* __restrict__ out)
{
    __shared__ _Float16 act[MT * ACT_STRIDE];   // 18.7 KB
    __shared__ float    pw[MT * 9][4];          // corner weights * mask
    __shared__ int      pidx[MT * 9][4];        // clamped gather indices

    const int tid  = threadIdx.x;
    const int blk  = blockIdx.x;
    const int wseg = blk & 7;
    const int h    = (blk >> 3) & (Hn - 1);
    const int b    = blk >> 10;
    const int w0   = wseg * 16;

    // --- phase 0: per (pixel, tap) bilinear parameters, mask folded in -----
    for (int t = tid; t < MT * 9; t += 128) {
        const int p = t / 9, k = t % 9;
        const int ky = k / 3 - 1, kx = k % 3 - 1;
        const int obase = (b * 9 + k) * HW + h * Wn + (w0 + p);
        const float ys = (float)(h + ky) + ybuf[obase];
        const float xs = (float)(w0 + p + kx) + xbuf[obase];
        const float m  = mbuf[obase];
        const float y0 = floorf(ys), x0 = floorf(xs);
        const float fy = ys - y0, fx = xs - x0;
#pragma unroll
        for (int c = 0; c < 4; ++c) {
            const int dy = c >> 1, dx = c & 1;
            const float yc = y0 + (float)dy, xc = x0 + (float)dx;
            const float wgt = (dy ? fy : 1.0f - fy) * (dx ? fx : 1.0f - fx);
            const bool ok = (yc >= 0.0f) && (yc <= (float)(Hn - 1)) &&
                            (xc >= 0.0f) && (xc <= (float)(Wn - 1));
            const int yi = min(max((int)yc, 0), Hn - 1);
            const int xi = min(max((int)xc, 0), Wn - 1);
            pw[t][c]   = ok ? wgt * m : 0.0f;
            pidx[t][c] = yi * Wn + xi;
        }
    }
    __syncthreads();

    // --- phase 1: gather mask*bilinear(x) into the f16 activation tile -----
    for (int e = tid; e < Cn * MT * 9; e += 128) {
        const int i  = e / (MT * 9);
        const int pk = e % (MT * 9);
        const int p = pk / 9, k = pk % 9;
        const float* xb = x + (size_t)(b * Cn + i) * HW;
        const float v = pw[pk][0] * xb[pidx[pk][0]] +
                        pw[pk][1] * xb[pidx[pk][1]] +
                        pw[pk][2] * xb[pidx[pk][2]] +
                        pw[pk][3] * xb[pidx[pk][3]];
        act[p * ACT_STRIDE + k * Cn + i] = (_Float16)v;
    }
    __syncthreads();

    // --- phase 2: 18 x v_wmma_f32_16x16x32_f16 ------------------------------
    const int wid = tid >> 5, lane = tid & 31;
    const _Float16* bp = Bp + (size_t)wid * KTILES * 32 * 16;
    v8f acc = wmma_gemm_576(act, bp, lane);

    // --- phase 3: epilogue (C fragment: VGPR r -> M=r or 8+r, lane -> N) ----
    const int o  = wid * 16 + (lane & 15);
    const int hi = (lane >> 4) & 1;
    const float bo = bias[o];
#pragma unroll
    for (int r = 0; r < 8; ++r) {
        const int p = r + hi * 8;
        const size_t oi = (size_t)(b * Cn + o) * HW + (size_t)h * Wn + (w0 + p);
        float v = acc[r] + bo;
        if (MODE == 0) v = (v >= 0.0f) ? v : 0.01f * v;   // leaky relu
        else           v += resid[oi];                    // residual add
        out[oi] = v;
    }
}

// ---------------------------------------------------------------------------
// Kernel 4: final 3x3 conv as the same M16 x N64 x K576 WMMA GEMM; phase 1 is
// a zero-padded im2col into the activation tile.
// ---------------------------------------------------------------------------
__global__ __launch_bounds__(128) void conv_gemm_kernel(
    const float* __restrict__ src, const _Float16* __restrict__ Bp,
    const float* __restrict__ bias, float* __restrict__ out)
{
    __shared__ _Float16 act[MT * ACT_STRIDE];

    const int tid  = threadIdx.x;
    const int blk  = blockIdx.x;
    const int wseg = blk & 7;
    const int h    = (blk >> 3) & (Hn - 1);
    const int b    = blk >> 10;
    const int w0   = wseg * 16;

    for (int e = tid; e < Cn * MT * 9; e += 128) {
        const int i  = e / (MT * 9);
        const int pk = e % (MT * 9);
        const int p = pk / 9, k = pk % 9;
        const int y = h + k / 3 - 1;
        const int xx = w0 + p + k % 3 - 1;
        float v = 0.0f;
        if (y >= 0 && y < Hn && xx >= 0 && xx < Wn)
            v = src[(size_t)(b * Cn + i) * HW + y * Wn + xx];
        act[p * ACT_STRIDE + k * Cn + i] = (_Float16)v;
    }
    __syncthreads();

    const int wid = tid >> 5, lane = tid & 31;
    const _Float16* bp = Bp + (size_t)wid * KTILES * 32 * 16;
    v8f acc = wmma_gemm_576(act, bp, lane);

    const int o  = wid * 16 + (lane & 15);
    const int hi = (lane >> 4) & 1;
    const float bo = bias[o];
#pragma unroll
    for (int r = 0; r < 8; ++r) {
        const int p = r + hi * 8;
        const size_t oi = (size_t)(b * Cn + o) * HW + (size_t)h * Wn + (w0 + p);
        out[oi] = acc[r] + bo;
    }
}

// ---------------------------------------------------------------------------
extern "C" void kernel_launch(void* const* d_in, const int* in_sizes, int n_in,
                              void* d_out, int out_size, void* d_ws, size_t ws_size,
                              hipStream_t stream)
{
    const float* input  = (const float*)d_in[0];
    const float* offset = (const float*)d_in[1];
    const float* w_off1 = (const float*)d_in[2];
    const float* b_off1 = (const float*)d_in[3];
    const float* w1     = (const float*)d_in[4];
    const float* b1     = (const float*)d_in[5];
    const float* w_off2 = (const float*)d_in[6];
    const float* b_off2 = (const float*)d_in[7];
    const float* w2     = (const float*)d_in[8];
    const float* b2     = (const float*)d_in[9];
    const float* w_last = (const float*)d_in[10];
    const float* b_last = (const float*)d_in[11];
    float* out = (float*)d_out;

    // Workspace carve-out (~41 MB total).
    char* ws = (char*)d_ws;
    auto carve = [&](size_t bytes) {
        char* p = ws;
        ws += (bytes + 255) & ~(size_t)255;
        return p;
    };
    const size_t offElems = (size_t)Bn * 9 * HW;
    float*    ybuf = (float*)carve(offElems * sizeof(float));
    float*    xbuf = (float*)carve(offElems * sizeof(float));
    float*    mbuf = (float*)carve(offElems * sizeof(float));
    float*    x1   = (float*)carve((size_t)Bn * Cn * HW * sizeof(float));
    float*    out2 = (float*)carve((size_t)Bn * Cn * HW * sizeof(float));
    const size_t packElems = (size_t)(Cn / 16) * KTILES * 32 * 16;
    _Float16* Bp1  = (_Float16*)carve(packElems * sizeof(_Float16));
    _Float16* Bp2  = (_Float16*)carve(packElems * sizeof(_Float16));
    _Float16* BpL  = (_Float16*)carve(packElems * sizeof(_Float16));
    (void)in_sizes; (void)n_in; (void)out_size; (void)ws_size;

    const int packBlocks = (int)((packElems + 255) / 256);  // 144
    pack_w_kernel<<<packBlocks, 256, 0, stream>>>(w1, Bp1);
    pack_w_kernel<<<packBlocks, 256, 0, stream>>>(w2, Bp2);
    pack_w_kernel<<<packBlocks, 256, 0, stream>>>(w_last, BpL);

    const int pixBlocks  = Bn * HW / 256;   // 256
    const int gemmBlocks = Bn * Hn * 8;     // 4096

    // DCN 1: offsets from w_off1, GEMM + leaky relu -> x1
    offset_conv_kernel<<<pixBlocks, 256, 0, stream>>>(offset, w_off1, b_off1,
                                                      ybuf, xbuf, mbuf);
    dcn_gemm_kernel<0><<<gemmBlocks, 128, 0, stream>>>(input, ybuf, xbuf, mbuf,
                                                       Bp1, b1, nullptr, x1);

    // DCN 2: offsets from w_off2 (reuse buffers), GEMM + residual(input) -> out2
    offset_conv_kernel<<<pixBlocks, 256, 0, stream>>>(offset, w_off2, b_off2,
                                                      ybuf, xbuf, mbuf);
    dcn_gemm_kernel<1><<<gemmBlocks, 128, 0, stream>>>(x1, ybuf, xbuf, mbuf,
                                                       Bp2, b2, input, out2);

    // Final 3x3 conv -> out
    conv_gemm_kernel<<<gemmBlocks, 128, 0, stream>>>(out2, BpL, b_last, out);
}
